// GraphAttentionLayer_31207232373166
// MI455X (gfx1250) — matrix operand order repaired and checked by reading, not compiled, not run on hardware
//
#include <hip/hip_runtime.h>
#include <hip/hip_bf16.h>

typedef __attribute__((ext_vector_type(16))) _Float16 v16h;
typedef __attribute__((ext_vector_type(8)))  float    v8f;

constexpr int kB = 8;
constexpr int kN = 2048;
constexpr int kF = 64;
constexpr float kSlope = 0.01f;

// ---- CDNA5 async global->LDS copy (16B per lane, tracked by ASYNCcnt) ----
typedef __attribute__((address_space(3))) const void lds_cv;
static __device__ __forceinline__ unsigned lds_off(const void* p) {
  return (unsigned)(unsigned long long)(lds_cv*)p;   // generic -> AS3 -> u32
}
#define ASYNC_CP16(ldsaddr, gptr, OFF)                                   \
  asm volatile("global_load_async_to_lds_b128 %0, %1, off offset:" OFF   \
               :: "v"(ldsaddr),                                          \
                  "v"((unsigned long long)(gptr)) : "memory")
static __device__ __forceinline__ void wait_async0() {
#if __has_builtin(__builtin_amdgcn_s_wait_asynccnt)
  __builtin_amdgcn_s_wait_asynccnt(0);
#else
  asm volatile("s_wait_asynccnt 0x0" ::: "memory");
#endif
}

// ---------------------------------------------------------------------------
// Kernel 1: per node j, compute Wh[j][:] (64 fp32), store transposed as f16
// WhT[b][f][j], plus f1[j] = Wh.a1 and f2[j] = Wh.a2 in fp32.
// ---------------------------------------------------------------------------
__global__ __launch_bounds__(64) void gat_prep(
    const float* __restrict__ h, const float* __restrict__ W,
    const float* __restrict__ av,
    _Float16* __restrict__ WhT, float* __restrict__ f1, float* __restrict__ f2)
{
  const int rowg = blockIdx.x;          // 0 .. kB*kN-1
  const int b    = rowg >> 11;          // / kN
  const int j    = rowg & (kN - 1);
  const int f    = threadIdx.x;         // 0..63

  __shared__ float hs[kF];
  hs[f] = h[(size_t)rowg * kF + f];
  __syncthreads();

  float acc = 0.f;
#pragma unroll
  for (int k = 0; k < kF; ++k) acc += hs[k] * W[k * kF + f];

  WhT[((size_t)b * kF + f) * kN + j] = (_Float16)acc;

  __shared__ float r1[kF];
  __shared__ float r2[kF];
  r1[f] = acc * av[f];
  r2[f] = acc * av[kF + f];
  __syncthreads();
#pragma unroll
  for (int s = 32; s > 0; s >>= 1) {
    if (f < s) { r1[f] += r1[f + s]; r2[f] += r2[f + s]; }
    __syncthreads();
  }
  if (f == 0) { f1[rowg] = r1[0]; f2[rowg] = r2[0]; }
}

// ---------------------------------------------------------------------------
// Kernel 2: per-batch max of f2. Exact softmax row max: leaky_relu is
// monotone, so max_j lrelu(f1_i + f2_j) = lrelu(f1_i + max_j f2_j).
// ---------------------------------------------------------------------------
__global__ __launch_bounds__(256) void gat_f2max(
    const float* __restrict__ f2, float* __restrict__ f2max)
{
  const int b = blockIdx.x;
  const int t = threadIdx.x;
  __shared__ float red[256];
  float m = -3.4e38f;
  for (int j = t; j < kN; j += 256) m = fmaxf(m, f2[b * kN + j]);
  red[t] = m;
  __syncthreads();
#pragma unroll
  for (int s = 128; s > 0; s >>= 1) {
    if (t < s) red[t] = fmaxf(red[t], red[t + s]);
    __syncthreads();
  }
  if (t == 0) f2max[b] = red[0];
}

// ---------------------------------------------------------------------------
// Kernel 3: fused GAT attention.
//   Block = 128 threads = 4 waves; wave w owns a 16-row tile (64 rows/block).
//   K-loop over j in chunks of 64 (two 32-j sub-chunks), double-buffered:
//     - each thread issues 4x global_load_async_to_lds_b128 (ASYNCcnt) to
//       stage the next 64x64 Wh chunk directly into LDS (no VGPR round-trip)
//     - each wave builds two P tiles = exp(lrelu(f1_i+f2_j) - m_i) in the
//       f16 WMMA A layout, then 10x v_wmma_f32_16x16x32_f16:
//       8 P@Wh tiles + 2 P@ones (row sums land in matching C layout)
//     - s_wait_asynccnt 0 + one barrier per 64 j's publishes the next buffer
//   Epilogue: out = C / Csum — no shuffles, no LDS reductions.
// ---------------------------------------------------------------------------
__global__ __launch_bounds__(128) void gat_main(
    const _Float16* __restrict__ WhT, const float* __restrict__ f1,
    const float* __restrict__ f2, const float* __restrict__ f2max,
    float* __restrict__ out)
{
  const int tid  = threadIdx.x;       // 0..127
  const int lane = tid & 31;
  const int wave = tid >> 5;          // 0..3
  const int b    = blockIdx.y;        // 0..kB-1
  const int i0   = blockIdx.x * 64 + wave * 16;
  const int row  = lane & 15;
  const bool hi  = lane >= 16;

  // Double-buffered 64-j chunk: [buf][sub][f][32 j's], 8KB/buffer, 16KB total.
  __shared__ _Float16 lds[2][2][kF * 32];

  const float f1r = f1[b * kN + i0 + row];
  float z = f1r + f2max[b];
  const float m = (z > 0.f) ? z : kSlope * z;   // exact per-row softmax max

  v8f c0 = {}; v8f c1 = {}; v8f c2 = {}; v8f c3 = {};
  v8f cs = {};                                  // row-sum accumulator
  v16h ones;
#pragma unroll
  for (int e = 0; e < 16; ++e) ones[e] = (_Float16)1.0f;

  const float* f2b   = f2 + b * kN;
  const int    koffA = hi ? 8 : 0;              // half the A-tile K offset (A uses k = 16h + s)
  const int    koffB = hi ? 16 : 0;             // B-tile K offset (this code's contiguous k = 16h + s)

  // Stager: thread tid owns feature row sf = tid/2, j-half sh = (tid&1)*16.
  const int sf = tid >> 1;
  const int sh = (tid & 1) * 16;
  const _Float16* const g0 = WhT + ((size_t)b * kF + sf) * kN + sh;
  const unsigned l0[2] = { lds_off(&lds[0][0][sf * 32 + sh]),
                           lds_off(&lds[1][0][sf * 32 + sh]) };
  const unsigned l1[2] = { lds_off(&lds[0][1][sf * 32 + sh]),
                           lds_off(&lds[1][1][sf * 32 + sh]) };

  constexpr int kChunks = kN / 64;

  // Prologue: async-stage chunk 0 into buffer 0 and publish it.
  {
    const unsigned long long g = (unsigned long long)g0;
    ASYNC_CP16(l0[0], g,      "0");
    ASYNC_CP16(l0[0], g,      "16");
    ASYNC_CP16(l1[0], g + 64, "0");   // +32 halves = sub-chunk 1
    ASYNC_CP16(l1[0], g + 64, "16");
  }
  wait_async0();
  __syncthreads();

  for (int ch = 0; ch < kChunks; ++ch) {
    const int jb  = ch * 64;
    const int buf = ch & 1;

    // ---- issue async staging of the NEXT chunk into the other buffer ----
    if (ch + 1 < kChunks) {
      const unsigned long long g =
          (unsigned long long)(g0 + (size_t)(ch + 1) * 64);
      ASYNC_CP16(l0[buf ^ 1], g,      "0");
      ASYNC_CP16(l0[buf ^ 1], g,      "16");
      ASYNC_CP16(l1[buf ^ 1], g + 64, "0");
      ASYNC_CP16(l1[buf ^ 1], g + 64, "16");
    }

    // ---- two sub-chunks of 32 j's: build A tile, 5 WMMAs each ----
#pragma unroll
    for (int s = 0; s < 2; ++s) {
      v16h a;
#pragma unroll
      for (int v = 0; v < 8; ++v) {
#pragma unroll
        for (int p = 0; p < 2; ++p) {
          const int K = (koffA << 1) + (v << 1) + p;
          float x = f1r + f2b[jb + s * 32 + K];
          x = fmaxf(x, kSlope * x);                // leaky_relu
          a[2 * v + p] = (_Float16)__expf(x - m);  // v_exp_f32
        }
      }

      const _Float16* const bsrc = &lds[buf][s][row * 32 + koffB];
      const v16h b0 = *(const v16h*)(bsrc + 0 * 512);
      const v16h b1 = *(const v16h*)(bsrc + 1 * 512);
      const v16h b2 = *(const v16h*)(bsrc + 2 * 512);
      const v16h b3 = *(const v16h*)(bsrc + 3 * 512);

      c0 = __builtin_amdgcn_wmma_f32_16x16x32_f16(false, a, false, b0,   (short)0, c0, false, false);
      c1 = __builtin_amdgcn_wmma_f32_16x16x32_f16(false, a, false, b1,   (short)0, c1, false, false);
      c2 = __builtin_amdgcn_wmma_f32_16x16x32_f16(false, a, false, b2,   (short)0, c2, false, false);
      c3 = __builtin_amdgcn_wmma_f32_16x16x32_f16(false, a, false, b3,   (short)0, c3, false, false);
      cs = __builtin_amdgcn_wmma_f32_16x16x32_f16(false, a, false, ones, (short)0, cs, false, false);
    }

    // ---- publish next buffer: my copies done + everyone's copies done ----
    if (ch + 1 < kChunks) {
      wait_async0();
      __syncthreads();
    }
  }

  // ---- normalize + store. C layout: lane<16 rows 0..7, lane>=16 rows 8..15,
  //      col = lane%16 + 16*tileIdx. cs[v] holds the matching row's sum. ----
  float* outr = out + ((size_t)b * kN + i0) * kF;
#pragma unroll
  for (int v = 0; v < 8; ++v) {
    const int r = v + (hi ? 8 : 0);
    const float inv = 1.0f / cs[v];
    outr[(size_t)r * kF + row +  0] = c0[v] * inv;
    outr[(size_t)r * kF + row + 16] = c1[v] * inv;
    outr[(size_t)r * kF + row + 32] = c2[v] * inv;
    outr[(size_t)r * kF + row + 48] = c3[v] * inv;
  }
}

// ---------------------------------------------------------------------------
// Launch: inputs (h, adj, W, a); adj is all-ones and unused by the reference.
// Workspace: WhT f16 (2 MB) + f1/f2 (128 KB) + 8 floats; fully rewritten
// every call (no cross-call state).
// ---------------------------------------------------------------------------
extern "C" void kernel_launch(void* const* d_in, const int* in_sizes, int n_in,
                              void* d_out, int out_size, void* d_ws, size_t ws_size,
                              hipStream_t stream) {
  const float* h  = (const float*)d_in[0];
  // d_in[1] = adj (unused by reference)
  const float* W  = (const float*)d_in[2];
  const float* av = (const float*)d_in[3];
  float* out = (float*)d_out;

  char* ws = (char*)d_ws;
  _Float16* WhT = (_Float16*)ws;
  float* f1    = (float*)(ws + (size_t)kB * kF * kN * sizeof(_Float16));
  float* f2    = f1 + (size_t)kB * kN;
  float* f2max = f2 + (size_t)kB * kN;

  gat_prep<<<kB * kN, 64, 0, stream>>>(h, W, av, WhT, f1, f2);
  gat_f2max<<<kB, 256, 0, stream>>>(f2, f2max);
  gat_main<<<dim3(kN / 64, kB), 128, 0, stream>>>(WhT, f1, f2, f2max, out);
}
